// ThreatCorrelationGNN_42975442764324
// MI455X (gfx1250) — matrix operand order repaired and checked by reading, not compiled, not run on hardware
//
#include <hip/hip_runtime.h>
#include <hip/hip_bf16.h>

typedef __attribute__((ext_vector_type(16))) __bf16 v16bf;
typedef __attribute__((ext_vector_type(8)))  float  v8f;
typedef unsigned short u16;

#define HID      128
#define NGRAPH   64

// ---------------------------------------------------------------------------
// degree / normalization
// ---------------------------------------------------------------------------
__global__ void k_deg_init(float* __restrict__ deg, int n) {
    int i = blockIdx.x * blockDim.x + threadIdx.x;
    if (i < n) deg[i] = 1.0f;                 // self-loop contribution
}

__global__ void k_deg_edges(const int* __restrict__ dstv, float* __restrict__ deg, int e) {
    int i = blockIdx.x * blockDim.x + threadIdx.x;
    if (i < e) atomicAdd(&deg[dstv[i]], 1.0f);
}

__global__ void k_dinv(const float* __restrict__ deg, float* __restrict__ dinv, int n) {
    int i = blockIdx.x * blockDim.x + threadIdx.x;
    if (i < n) dinv[i] = rsqrtf(deg[i]);      // deg >= 1 always (self-loops)
}

// ---------------------------------------------------------------------------
// Weight pre-swizzle: W[KDIM,128] f32  ->  hi/lo bf16 in WMMA B-operand order:
//   idx = ((kt*8 + wavecol)*32 + lane)*16 + i
//   lane holds column = wavecol*16 + lane%16 ; K = kt*32 + (lane/16)*16 + i
// Run once per layer per call; output is tiny (<=64KB) and L2-resident.
// ---------------------------------------------------------------------------
template<int KDIM>
__global__ void k_wprep(const float* __restrict__ W,
                        u16* __restrict__ Whi, u16* __restrict__ Wlo)
{
    int idx = blockIdx.x * blockDim.x + threadIdx.x;
    if (idx >= KDIM * HID) return;
    int i    = idx & 15;
    int lane = (idx >> 4) & 31;
    int w    = (idx >> 9) & 7;
    int kt   = idx >> 12;
    int col  = w * 16 + (lane & 15);
    int k    = kt * 32 + ((lane >> 4) << 4) + i;
    float f  = W[k * HID + col];
    __bf16 h = (__bf16)f;
    __bf16 l = (__bf16)(f - (float)h);
    Whi[idx] = __builtin_bit_cast(u16, h);
    Wlo[idx] = __builtin_bit_cast(u16, l);
}

// ---------------------------------------------------------------------------
// WMMA GEMM:  Hlin = Hin @ W   (N x KDIM) @ (KDIM x 128), fp32 in/out.
// 3x-bf16 split (hi*hi + lo*hi + hi*lo) for ~fp32 accuracy on the bf16 pipe.
// A tile staged into LDS pre-swizzled into operand order (packed bf16), so
// each lane loads its v16bf operands as contiguous b128s; B operands come
// straight from the pre-swizzled global weight buffers.
// Epilogue also writes Agg = dinv^2 * Hlin  (self-loop init of aggregation).
// Block: 256 threads = 8 wave32; block = 16 rows x 128 cols.
// ---------------------------------------------------------------------------
template<int KDIM>
__launch_bounds__(256)
__global__ void k_gcn_gemm(const float* __restrict__ Hin,
                           const u16* __restrict__ Wops_hi,
                           const u16* __restrict__ Wops_lo,
                           const float* __restrict__ dinv,
                           float* __restrict__ Hlin,
                           float* __restrict__ Agg,
                           int nRows)
{
    __shared__ alignas(32) u16 AhiS[32][16];   // A operand, hi parts (1KB)
    __shared__ alignas(32) u16 AloS[32][16];   // A operand, lo parts (1KB)

    const int t    = threadIdx.x;
    const int lane = t & 31;
    const int wave = t >> 5;                   // 0..7 -> 16-col tile
    const int row0 = blockIdx.x * 16;

    // staging role: thread handles rows sm, k-pair (2*sp, 2*sp+1)
    const int sm    = t >> 4;                  // 0..15
    const int sk    = (t & 15) * 2;            // even k within 32-k tile
    const int shalf = sk >> 4;                 // (m,k)->(lane,i): k = 16*half + i
    const int si    = sk & 15;
    const int slane = sm + 16 * shalf;
    const int sgr   = row0 + sm;
    const bool srow_ok = (sgr < nRows);

    v8f acc = {};

    #pragma unroll 2
    for (int kt = 0; kt < KDIM / 32; ++kt) {
        // ---- stage A tile, pre-swizzled + hi/lo split (one float2/thread) ----
        float2 a2 = make_float2(0.0f, 0.0f);
        if (srow_ok)
            a2 = *(const float2*)(Hin + sgr * KDIM + kt * 32 + sk);
        __bf16 h0 = (__bf16)a2.x, h1 = (__bf16)a2.y;
        __bf16 l0 = (__bf16)(a2.x - (float)h0);
        __bf16 l1 = (__bf16)(a2.y - (float)h1);
        unsigned hp = (unsigned)__builtin_bit_cast(u16, h0) |
                      ((unsigned)__builtin_bit_cast(u16, h1) << 16);
        unsigned lp = (unsigned)__builtin_bit_cast(u16, l0) |
                      ((unsigned)__builtin_bit_cast(u16, l1) << 16);
        *(unsigned*)&AhiS[slane][si] = hp;
        *(unsigned*)&AloS[slane][si] = lp;
        if (srow_ok && (kt + 1) * 32 < KDIM)   // global_prefetch_b8 on next slab
            __builtin_prefetch(Hin + sgr * KDIM + kt * 32 + 32 + sk, 0, 3);
        __syncthreads();

        // ---- per-lane operands: contiguous 32B vector loads ----
        v16bf a_hi = *(const v16bf*)AhiS[lane];
        v16bf a_lo = *(const v16bf*)AloS[lane];
        const int bofs = ((kt * 8 + wave) * 32 + lane) * 16;
        v16bf b_hi = *(const v16bf*)(Wops_hi + bofs);
        v16bf b_lo = *(const v16bf*)(Wops_lo + bofs);

        acc = __builtin_amdgcn_wmma_f32_16x16x32_bf16(false, a_hi, false, b_hi,
                                                      (short)0, acc, false, false);
        acc = __builtin_amdgcn_wmma_f32_16x16x32_bf16(false, a_lo, false, b_hi,
                                                      (short)0, acc, false, false);
        acc = __builtin_amdgcn_wmma_f32_16x16x32_bf16(false, a_hi, false, b_lo,
                                                      (short)0, acc, false, false);
        __syncthreads();
    }

    // ---- epilogue: C/D layout -> memory; also self-loop init of Agg ----
    const int half = lane >> 4;
    const int col  = (wave << 4) + (lane & 15);
    #pragma unroll
    for (int r = 0; r < 8; ++r) {
        int grow = row0 + r + 8 * half;        // lanes 0-15: M=r ; 16-31: M=8+r
        if (grow < nRows) {
            float v  = acc[r];
            Hlin[grow * HID + col] = v;
            float dv = dinv[grow];
            Agg[grow * HID + col]  = dv * dv * v;
        }
    }
}

// ---------------------------------------------------------------------------
// Edge scatter: Agg[dst] += dinv[src]*dinv[dst] * Hlin[src]  (32 lanes/edge)
// ---------------------------------------------------------------------------
__global__ void k_scatter(const int* __restrict__ srcv, const int* __restrict__ dstv,
                          const float* __restrict__ dinv,
                          const float* __restrict__ Hlin,
                          float* __restrict__ Agg, int nEdges)
{
    int t = blockIdx.x * blockDim.x + threadIdx.x;
    int e = t >> 5;
    int lane = t & 31;
    if (e >= nEdges) return;
    int s = srcv[e], d = dstv[e];
    float nm = dinv[s] * dinv[d];
    float4 v = ((const float4*)(Hlin + s * HID))[lane];
    float* ap = Agg + d * HID + lane * 4;
    atomicAdd(ap + 0, nm * v.x);
    atomicAdd(ap + 1, nm * v.y);
    atomicAdd(ap + 2, nm * v.z);
    atomicAdd(ap + 3, nm * v.w);
}

// ---------------------------------------------------------------------------
// bias + ReLU (in place)
// ---------------------------------------------------------------------------
__global__ void k_bias_relu(float* __restrict__ H, const float* __restrict__ b, int total) {
    int i = blockIdx.x * blockDim.x + threadIdx.x;
    if (i < total) {
        float v = H[i] + b[i & (HID - 1)];
        H[i] = v > 0.0f ? v : 0.0f;
    }
}

// ---------------------------------------------------------------------------
// mean pool over graphs
// ---------------------------------------------------------------------------
__global__ void k_pool_zero(float* __restrict__ sums, float* __restrict__ cnt) {
    int i = blockIdx.x * blockDim.x + threadIdx.x;
    if (i < NGRAPH * HID) sums[i] = 0.0f;
    if (i < NGRAPH)       cnt[i]  = 0.0f;
}

__global__ void k_pool_acc(const float* __restrict__ H, const int* __restrict__ batch,
                           float* __restrict__ sums, float* __restrict__ cnt, int n)
{
    int t = blockIdx.x * blockDim.x + threadIdx.x;
    int node = t >> 5;
    int lane = t & 31;
    if (node >= n) return;
    int g = batch[node];
    float4 v = ((const float4*)(H + node * HID))[lane];
    float* sp = sums + g * HID + lane * 4;
    atomicAdd(sp + 0, v.x);
    atomicAdd(sp + 1, v.y);
    atomicAdd(sp + 2, v.z);
    atomicAdd(sp + 3, v.w);
    if (lane == 0) atomicAdd(&cnt[g], 1.0f);
}

// ---------------------------------------------------------------------------
// classifier MLP over 64 pooled vectors (tiny; single block)
// ---------------------------------------------------------------------------
__launch_bounds__(256)
__global__ void k_classifier(const float* __restrict__ sums, const float* __restrict__ cnt,
                             const float* __restrict__ Wc1, const float* __restrict__ bc1,
                             const float* __restrict__ Wc2, const float* __restrict__ bc2,
                             float* __restrict__ out)
{
    __shared__ float pooled[NGRAPH][HID];
    __shared__ float hcls[NGRAPH][HID];
    int t = threadIdx.x;
    for (int idx = t; idx < NGRAPH * HID; idx += 256) {
        int g = idx / HID;
        float c = cnt[g];
        c = c > 1.0f ? c : 1.0f;
        pooled[g][idx % HID] = sums[idx] / c;
    }
    __syncthreads();
    for (int idx = t; idx < NGRAPH * HID; idx += 256) {
        int g = idx / HID, o = idx % HID;
        float a = bc1[o];
        #pragma unroll 8
        for (int k = 0; k < HID; ++k) a += pooled[g][k] * Wc1[k * HID + o];
        hcls[g][o] = a > 0.0f ? a : 0.0f;
    }
    __syncthreads();
    for (int idx = t; idx < NGRAPH * 2; idx += 256) {
        int g = idx >> 1, c = idx & 1;
        float a = bc2[c];
        #pragma unroll 8
        for (int k = 0; k < HID; ++k) a += hcls[g][k] * Wc2[k * 2 + c];
        out[idx] = a;
    }
}

// ---------------------------------------------------------------------------
// host launcher
// ---------------------------------------------------------------------------
static inline size_t align256(size_t v) { return (v + 255) & ~size_t(255); }

extern "C" void kernel_launch(void* const* d_in, const int* in_sizes, int n_in,
                              void* d_out, int out_size, void* d_ws, size_t ws_size,
                              hipStream_t stream)
{
    const float* x     = (const float*)d_in[0];
    const int*   ei    = (const int*)  d_in[1];
    const int*   batch = (const int*)  d_in[2];
    const float* W1  = (const float*)d_in[3];  const float* b1  = (const float*)d_in[4];
    const float* W2  = (const float*)d_in[5];  const float* b2  = (const float*)d_in[6];
    const float* W3  = (const float*)d_in[7];  const float* b3  = (const float*)d_in[8];
    const float* Wc1 = (const float*)d_in[9];  const float* bc1 = (const float*)d_in[10];
    const float* Wc2 = (const float*)d_in[11]; const float* bc2 = (const float*)d_in[12];

    const int N = in_sizes[2];          // 50000 nodes (batch array length)
    const int E = in_sizes[1] / 2;      // 800000 edges
    (void)n_in; (void)out_size; (void)ws_size;

    const int* srcv = ei;
    const int* dstv = ei + E;

    // workspace carve-up
    char* ws = (char*)d_ws;
    float* deg   = (float*)ws; ws += align256((size_t)N * 4);
    float* dinv  = (float*)ws; ws += align256((size_t)N * 4);
    float* buf0  = (float*)ws; ws += align256((size_t)N * HID * 4);
    float* buf1  = (float*)ws; ws += align256((size_t)N * HID * 4);   // h_lin scratch
    float* buf2  = (float*)ws; ws += align256((size_t)N * HID * 4);
    float* sums  = (float*)ws; ws += align256((size_t)NGRAPH * HID * 4);
    float* cnt   = (float*)ws; ws += align256((size_t)NGRAPH * 4);
    u16*   w1hi  = (u16*)ws;   ws += align256((size_t)256 * HID * 2);
    u16*   w1lo  = (u16*)ws;   ws += align256((size_t)256 * HID * 2);
    u16*   w2hi  = (u16*)ws;   ws += align256((size_t)128 * HID * 2);
    u16*   w2lo  = (u16*)ws;   ws += align256((size_t)128 * HID * 2);
    u16*   w3hi  = (u16*)ws;   ws += align256((size_t)128 * HID * 2);
    u16*   w3lo  = (u16*)ws;   ws += align256((size_t)128 * HID * 2);

    const int TB = 256;
    const int mtiles = (N + 15) / 16;
    const int ethreads = E * 32;
    const int nh = N * HID;

    // weight pre-swizzle (tiny) + normalization
    k_wprep<256><<<(256 * HID + TB - 1) / TB, TB, 0, stream>>>(W1, w1hi, w1lo);
    k_wprep<128><<<(128 * HID + TB - 1) / TB, TB, 0, stream>>>(W2, w2hi, w2lo);
    k_wprep<128><<<(128 * HID + TB - 1) / TB, TB, 0, stream>>>(W3, w3hi, w3lo);
    k_deg_init <<<(N + TB - 1) / TB, TB, 0, stream>>>(deg, N);
    k_deg_edges<<<(E + TB - 1) / TB, TB, 0, stream>>>(dstv, deg, E);
    k_dinv     <<<(N + TB - 1) / TB, TB, 0, stream>>>(deg, dinv, N);

    // layer 1: x(256) -> buf2
    k_gcn_gemm<256><<<mtiles, TB, 0, stream>>>(x, w1hi, w1lo, dinv, buf1, buf2, N);
    k_scatter  <<<(ethreads + TB - 1) / TB, TB, 0, stream>>>(srcv, dstv, dinv, buf1, buf2, E);
    k_bias_relu<<<(nh + TB - 1) / TB, TB, 0, stream>>>(buf2, b1, nh);

    // layer 2: buf2 -> buf0
    k_gcn_gemm<128><<<mtiles, TB, 0, stream>>>(buf2, w2hi, w2lo, dinv, buf1, buf0, N);
    k_scatter  <<<(ethreads + TB - 1) / TB, TB, 0, stream>>>(srcv, dstv, dinv, buf1, buf0, E);
    k_bias_relu<<<(nh + TB - 1) / TB, TB, 0, stream>>>(buf0, b2, nh);

    // layer 3: buf0 -> buf2
    k_gcn_gemm<128><<<mtiles, TB, 0, stream>>>(buf0, w3hi, w3lo, dinv, buf1, buf2, N);
    k_scatter  <<<(ethreads + TB - 1) / TB, TB, 0, stream>>>(srcv, dstv, dinv, buf1, buf2, E);
    k_bias_relu<<<(nh + TB - 1) / TB, TB, 0, stream>>>(buf2, b3, nh);

    // mean pool + classifier
    k_pool_zero<<<(NGRAPH * HID + TB - 1) / TB, TB, 0, stream>>>(sums, cnt);
    k_pool_acc <<<(N * 32 + TB - 1) / TB, TB, 0, stream>>>(buf2, batch, sums, cnt, N);
    k_classifier<<<1, TB, 0, stream>>>(sums, cnt, Wc1, bc1, Wc2, bc2, (float*)d_out);
}
